// NIE_GCN_50818053046990
// MI455X (gfx1250) — matrix operand order repaired and checked by reading, not compiled, not run on hardware
//
#include <hip/hip_runtime.h>
#include <hip/hip_bf16.h>
#include <math.h>

typedef __attribute__((ext_vector_type(2)))  float    v2f;
typedef __attribute__((ext_vector_type(8)))  float    v8f;
typedef __attribute__((ext_vector_type(16))) _Float16 v16h;

#define DIM  64
#define FEAT 512

// ---------------------------------------------------------------- zero fill
__global__ void nie_zero_f32(float* __restrict__ p, int n) {
  int i = blockIdx.x * blockDim.x + threadIdx.x;
  if (i < n) p[i] = 0.f;
}

// ---------------------------------------------------------------- projection
// emb[rows, 64] = sim[rows, 512] @ W^T  (W stored [64, 512] row-major)
// One wave per 16x16 output tile, fp32 WMMA 16x16x4, K-loop of 128 steps.
__global__ void nie_proj_wmma(const float* __restrict__ sim,
                              const float* __restrict__ W,
                              float* __restrict__ emb, int mtiles) {
  int wave = (blockIdx.x * blockDim.x + threadIdx.x) >> 5;
  int lane = threadIdx.x & 31;
  if (wave >= mtiles * 4) return;
  int mt = wave >> 2, nt = wave & 3;
  int h = lane >> 4, L = lane & 15;

  const float* arow = sim + (size_t)(mt * 16 + L) * FEAT;  // A row M = L
  const float* brow = W   + (size_t)(nt * 16 + L) * FEAT;  // B col N = L (B=W^T)

  v8f c = {0.f, 0.f, 0.f, 0.f, 0.f, 0.f, 0.f, 0.f};
  for (int ks = 0; ks < FEAT / 4; ++ks) {
    int kb = ks * 4 + 2 * h;               // f32 A/B layout: lanes 16-31 hold K+2
    v2f a = {arow[kb], arow[kb + 1]};
    v2f b = {brow[kb], brow[kb + 1]};
    c = __builtin_amdgcn_wmma_f32_16x16x4_f32(false, a, false, b,
                                              (short)0, c, false, false);
  }
#pragma unroll
  for (int v = 0; v < 8; ++v)              // D: VGPR v -> M = v + 8h, N = L
    emb[(size_t)(mt * 16 + v + 8 * h) * DIM + nt * 16 + L] = c[v];
}

// ---------------------------------------------------------------- edge attention
// Per 16-edge tile (one wave): e = relu([m_emb[mi] | d_emb[di]]) (16x128),
// h = tanh(e @ W1^T + b1) via f16 WMMA 16x16x32, score = h @ W2, w = exp(score).
// Atomically accumulates rowsums and degree counts.
__global__ void nie_edge_attn(const float* __restrict__ m_emb,
                              const float* __restrict__ d_emb,
                              const float* __restrict__ W1,
                              const float* __restrict__ b1,
                              const float* __restrict__ W2,
                              const int* __restrict__ mi,
                              const int* __restrict__ di,
                              float* __restrict__ wbuf,
                              float* __restrict__ rowsum_m,
                              float* __restrict__ rowsum_d,
                              int* __restrict__ cnt_m,
                              int* __restrict__ cnt_d, int ntiles) {
  __shared__ _Float16 w1h[128 * 64];         // W1^T as f16, [k][n]
  __shared__ float    b1s[64], w2s[64];
  __shared__ _Float16 atile[8][16 * 128];    // per-wave A tiles

  int tid = threadIdx.x;
  for (int i = tid; i < 128 * 64; i += 256) {
    int k = i >> 6, n = i & 63;
    w1h[i] = (_Float16)W1[n * 128 + k];
  }
  if (tid < 64) { b1s[tid] = b1[tid]; w2s[tid] = W2[tid]; }
  __syncthreads();

  int wave = tid >> 5, lane = tid & 31;
  int tile = blockIdx.x * 8 + wave;
  if (tile >= ntiles) return;

  _Float16* at = atile[wave];
  int r  = lane & 15;                        // edge slot within tile
  int eg = tile * 16 + r;
  bool hi = lane >= 16;                      // lanes 16-31 load the d-half
  int node = hi ? di[eg] : mi[eg];
  const float* rowp = (hi ? d_emb : m_emb) + (size_t)node * DIM;
  int koff = hi ? 64 : 0;
  for (int k = 0; k < 64; k += 4) {
    float4 v = *reinterpret_cast<const float4*>(rowp + k);
    at[r * 128 + koff + k + 0] = (_Float16)fmaxf(v.x, 0.f);
    at[r * 128 + koff + k + 1] = (_Float16)fmaxf(v.y, 0.f);
    at[r * 128 + koff + k + 2] = (_Float16)fmaxf(v.z, 0.f);
    at[r * 128 + koff + k + 3] = (_Float16)fmaxf(v.w, 0.f);
  }
  // within-wave DS ops are in-order: LDS writes above are visible to reads below

  int h = lane >> 4, Mrow = lane & 15;       // Mrow doubles as N for B / C
  v8f c[4];
#pragma unroll
  for (int t = 0; t < 4; ++t) c[t] = (v8f){0.f,0.f,0.f,0.f,0.f,0.f,0.f,0.f};

  for (int ks = 0; ks < 4; ++ks) {           // K = 128 in 4 steps of 32
    v16h a;
#pragma unroll
    for (int v = 0; v < 8; ++v) {            // f16 A layout (16x32)
      int kb = ks * 32 + h * 16 + 2 * v;
      a[2 * v]     = at[Mrow * 128 + kb];
      a[2 * v + 1] = at[Mrow * 128 + kb + 1];
    }
#pragma unroll
    for (int t = 0; t < 4; ++t) {
      v16h b;
#pragma unroll
      for (int v = 0; v < 8; ++v) {          // f16 B layout (32x16)
        int kb = ks * 32 + h * 16 + 2 * v;
        int n  = t * 16 + Mrow;
        b[2 * v]     = w1h[kb * 64 + n];
        b[2 * v + 1] = w1h[(kb + 1) * 64 + n];
      }
      c[t] = __builtin_amdgcn_wmma_f32_16x16x32_f16(false, a, false, b,
                                                    (short)0, c[t], false, false);
    }
  }

  // tanh, dot with W2, reduce over n (16-lane groups), exp, scatter
  float sv[8];
#pragma unroll
  for (int v = 0; v < 8; ++v) sv[v] = 0.f;
#pragma unroll
  for (int t = 0; t < 4; ++t) {
    int n = t * 16 + Mrow;
    float bb = b1s[n], w2v = w2s[n];
#pragma unroll
    for (int v = 0; v < 8; ++v)
      sv[v] += tanhf(c[t][v] + bb) * w2v;
  }
#pragma unroll
  for (int v = 0; v < 8; ++v) {
    sv[v] += __shfl_xor(sv[v], 8, 16);
    sv[v] += __shfl_xor(sv[v], 4, 16);
    sv[v] += __shfl_xor(sv[v], 2, 16);
    sv[v] += __shfl_xor(sv[v], 1, 16);
  }
  if (Mrow == 0) {                            // lanes 0 and 16: 8 edges each
#pragma unroll
    for (int v = 0; v < 8; ++v) {
      int e2 = tile * 16 + v + 8 * h;         // C: VGPR v -> edge v + 8h
      float wv = expf(sv[v]);
      wbuf[e2] = wv;
      int rm = mi[e2], rd = di[e2];
      atomicAdd(&rowsum_m[rm], wv);
      atomicAdd(&rowsum_d[rd], wv);
      atomicAdd(&cnt_m[rm], 1);
      atomicAdd(&cnt_d[rd], 1);
    }
  }
}

// ---------------------------------------------------------------- exclusive scan
__global__ void nie_scan_excl(const int* __restrict__ cnt, int* __restrict__ off,
                              int* __restrict__ cur, int n) {
  __shared__ int csum[256];
  int tid = threadIdx.x;
  int chunk = (n + 255) / 256;
  int s = tid * chunk;
  int e = s + chunk; if (e > n) e = n;
  int acc = 0;
  for (int i = s; i < e; ++i) acc += cnt[i];
  csum[tid] = acc;
  __syncthreads();
  if (tid == 0) {
    int run = 0;
    for (int j = 0; j < 256; ++j) { int t = csum[j]; csum[j] = run; run += t; }
    off[n] = run;
  }
  __syncthreads();
  int run = csum[tid];
  for (int i = s; i < e; ++i) { off[i] = run; cur[i] = run; run += cnt[i]; }
}

// ---------------------------------------------------------------- CSR scatter + norm
__global__ void nie_scatter_norm(const int* __restrict__ mi, const int* __restrict__ di,
                                 const float* __restrict__ w,
                                 const float* __restrict__ rsm, const float* __restrict__ rsd,
                                 int* __restrict__ cur_m, int* __restrict__ cur_d,
                                 int* __restrict__ elm, int* __restrict__ eld,
                                 float* __restrict__ wm, float* __restrict__ wd, int E) {
  int e = blockIdx.x * blockDim.x + threadIdx.x;
  if (e >= E) return;
  int a = mi[e], b = di[e];
  float wv = w[e];
  wm[e] = wv * rsqrtf(rsm[a]);   // row has >=1 edge => rowsum >= wv > 0
  wd[e] = wv * rsqrtf(rsd[b]);
  int pm = atomicAdd(&cur_m[a], 1); elm[pm] = e;
  int pd = atomicAdd(&cur_d[b], 1); eld[pd] = e;
}

// ---------------------------------------------------------------- sparse GCN pass
// One wave per output row: dst[row] = tanh(sum_e wn[e]*src[oidx[e]]); acc += dst
__global__ void nie_spmm_tanh(const int* __restrict__ off, const int* __restrict__ elist,
                              const int* __restrict__ oidx, const float* __restrict__ wn,
                              const float* __restrict__ src, float* __restrict__ dst,
                              float* __restrict__ acc, int rows) {
  int row  = (blockIdx.x * blockDim.x + threadIdx.x) >> 5;
  int lane = threadIdx.x & 31;
  if (row >= rows) return;
  int beg = off[row], end = off[row + 1];
  float s0 = 0.f, s1 = 0.f;
  for (int j = beg; j < end; ++j) {
    __builtin_prefetch(&elist[j + 64], 0, 0);   // stream edge ids ahead
    int e = elist[j];
    float wgt = wn[e];
    const float* sp = src + (size_t)oidx[e] * DIM;
    s0 += wgt * sp[lane];
    s1 += wgt * sp[lane + 32];
  }
  float t0 = tanhf(s0), t1 = tanhf(s1);
  size_t base = (size_t)row * DIM + lane;
  dst[base] = t0;       dst[base + 32] = t1;
  acc[base] += t0;      acc[base + 32] += t1;
}

// ---------------------------------------------------------------- launcher
extern "C" void kernel_launch(void* const* d_in, const int* in_sizes, int n_in,
                              void* d_out, int out_size, void* d_ws, size_t ws_size,
                              hipStream_t stream) {
  const float* m_sim = (const float*)d_in[0];
  const float* d_sim = (const float*)d_in[1];
  const float* Wm    = (const float*)d_in[2];
  const float* Wd    = (const float*)d_in[3];
  const float* W1    = (const float*)d_in[4];
  const float* b1    = (const float*)d_in[5];
  const float* W2    = (const float*)d_in[6];
  const int*   mi    = (const int*)d_in[7];
  const int*   di    = (const int*)d_in[8];
  const int NM = in_sizes[0] / FEAT;   // 12000
  const int ND = in_sizes[1] / FEAT;   // 10000
  const int E  = in_sizes[7];          // 1,000,000

  // workspace carve-out (256B-aligned blocks)
  char* ws = (char*)d_ws;
  size_t o = 0;
  auto take = [&](size_t bytes) -> void* {
    o = (o + 255) & ~(size_t)255;
    void* p = ws + o;
    o += bytes;
    return p;
  };
  // meta block kept contiguous so one zero-fill covers it
  float* rowsum_m = (float*)take((size_t)(NM + ND) * 2 * 4);
  float* rowsum_d = rowsum_m + NM;
  int*   cnt_m    = (int*)(rowsum_d + ND);
  int*   cnt_d    = cnt_m + NM;
  float* m_emb = (float*)take((size_t)NM * DIM * 4);
  float* d_emb = (float*)take((size_t)ND * DIM * 4);
  float* wbuf  = (float*)take((size_t)E * 4);
  float* wm    = (float*)take((size_t)E * 4);
  float* wd    = (float*)take((size_t)E * 4);
  int*   off_m = (int*)take((size_t)(NM + 1) * 4);
  int*   off_d = (int*)take((size_t)(ND + 1) * 4);
  int*   cur_m = (int*)take((size_t)NM * 4);
  int*   cur_d = (int*)take((size_t)ND * 4);
  int*   elm   = (int*)take((size_t)E * 4);
  int*   eld   = (int*)take((size_t)E * 4);
  (void)ws_size; (void)n_in;

  // 1. zero meta + output accumulators (harness poisons both)
  int metaWords = (NM + ND) * 2;
  nie_zero_f32<<<(metaWords + 255) / 256, 256, 0, stream>>>(rowsum_m, metaWords);
  nie_zero_f32<<<(out_size + 255) / 256, 256, 0, stream>>>((float*)d_out, out_size);

  // 2. fp32 WMMA projections
  int mtM = NM / 16, mtD = ND / 16;
  nie_proj_wmma<<<(mtM * 4 * 32 + 255) / 256, 256, 0, stream>>>(m_sim, Wm, m_emb, mtM);
  nie_proj_wmma<<<(mtD * 4 * 32 + 255) / 256, 256, 0, stream>>>(d_sim, Wd, d_emb, mtD);

  // 3. f16 WMMA edge attention + rowsums + degree histogram
  int etiles = E / 16;
  nie_edge_attn<<<(etiles + 7) / 8, 256, 0, stream>>>(
      m_emb, d_emb, W1, b1, W2, mi, di, wbuf, rowsum_m, rowsum_d, cnt_m, cnt_d, etiles);

  // 4. CSR build + beta=0.5 row normalization
  nie_scan_excl<<<1, 256, 0, stream>>>(cnt_m, off_m, cur_m, NM);
  nie_scan_excl<<<1, 256, 0, stream>>>(cnt_d, off_d, cur_d, ND);
  nie_scatter_norm<<<(E + 255) / 256, 256, 0, stream>>>(
      mi, di, wbuf, rowsum_m, rowsum_d, cur_m, cur_d, elm, eld, wm, wd, E);

  // 5. three GCN layers as sparse gather-SpMM (L2-resident embeddings)
  float* acc_m = (float*)d_out;
  float* acc_d = acc_m + (size_t)NM * DIM;
  for (int l = 0; l < 3; ++l) {
    nie_spmm_tanh<<<(NM * 32 + 255) / 256, 256, 0, stream>>>(
        off_m, elm, di, wm, d_emb, m_emb, acc_m, NM);
    nie_spmm_tanh<<<(ND * 32 + 255) / 256, 256, 0, stream>>>(
        off_d, eld, mi, wd, m_emb, d_emb, acc_d, ND);
  }
}
